// GlobalAttention_15736760172825
// MI455X (gfx1250) — matrix operand order repaired and checked by reading, not compiled, not run on hardware
//
#include <hip/hip_runtime.h>
#include <math.h>

// Problem constants (from reference)
#define B_   32
#define TGT_ 512
#define SRC_ 2048
#define H_   512
#define H2_  1024

typedef _Float16 h16x16 __attribute__((ext_vector_type(16)));
typedef _Float16 h16x8  __attribute__((ext_vector_type(8)));
typedef float    f32x8  __attribute__((ext_vector_type(8)));

// ---------------------------------------------------------------------------
// CDNA5 async global->LDS copy (ASYNCcnt-tracked), with portable fallback
// Builtin signature (from hipcc diagnostic): param0 is int4* in the global
// (__device__/AS1) address space; param1 is the LDS-side pointer.
// ---------------------------------------------------------------------------
#if defined(__has_builtin)
#if __has_builtin(__builtin_amdgcn_global_load_async_to_lds_b128)
#define HAVE_ASYNC_LDS 1
#endif
#endif
#ifndef HAVE_ASYNC_LDS
#define HAVE_ASYNC_LDS 0
#endif

typedef int i32x4_ __attribute__((vector_size(16)));
typedef i32x4_ __attribute__((address_space(1)))* as1_i32x4_p;
typedef i32x4_ __attribute__((address_space(3)))* as3_i32x4_p;

__device__ __forceinline__ void cp_async16(const _Float16* g, _Float16* l) {
#if HAVE_ASYNC_LDS
  __builtin_amdgcn_global_load_async_to_lds_b128(
      (as1_i32x4_p)(g), (as3_i32x4_p)(l), 0, 0);
#else
  *(h16x8*)l = *(const h16x8*)g;
#endif
}

__device__ __forceinline__ void wait_async_and_barrier() {
#if HAVE_ASYNC_LDS
#if __has_builtin(__builtin_amdgcn_s_wait_asynccnt)
  __builtin_amdgcn_s_wait_asynccnt(0);
#else
  asm volatile("s_wait_asynccnt 0" ::: "memory");
#endif
#endif
  __syncthreads();
}

// ---------------------------------------------------------------------------
// LDS tile geometry: block tile 64(M) x 128(N), K-step 32, double-buffered.
// Padded strides (in halfs) keep rows 16B-aligned while skewing banks.
// ---------------------------------------------------------------------------
#define LDS_A_STRIDE 40   // 32 + 8
#define LDS_B_STRIDE 136  // 128 + 8

// stage A tile: 64 rows x 32 halfs; one b128 chunk per thread (256 threads)
__device__ __forceinline__ void stage_A_f16(_Float16* dst, const _Float16* src,
                                            int lda, int tid) {
  int m = tid >> 2, kc = (tid & 3) << 3;
  cp_async16(src + (size_t)m * lda + kc, dst + m * LDS_A_STRIDE + kc);
}

// stage A tile from fp32 source with f32->f16 conversion (register path)
__device__ __forceinline__ void stage_A_f32(_Float16* dst, const float* src,
                                            long lda, int tid) {
  int m = tid >> 2, kc = (tid & 3) << 3;
  f32x8 v = *(const f32x8*)(src + (long)m * lda + kc);
  h16x8 h;
#pragma unroll
  for (int i = 0; i < 8; ++i) h[i] = (_Float16)v[i];
  *(h16x8*)(dst + m * LDS_A_STRIDE + kc) = h;
}

// stage B tile: 32 rows x 128 halfs; two b128 chunks per thread
__device__ __forceinline__ void stage_B_f16(_Float16* dst, const _Float16* src,
                                            int ldb, int tid) {
  int r = tid >> 3, nc = (tid & 7) << 3;
  cp_async16(src + (size_t)r * ldb + nc,      dst + r * LDS_B_STRIDE + nc);
  cp_async16(src + (size_t)r * ldb + nc + 64, dst + r * LDS_B_STRIDE + nc + 64);
}

// ---------------------------------------------------------------------------
// WMMA fragments from LDS (V_WMMA_F32_16X16X32_F16 layouts)
// A (16x32 MxK): lane L: m=L&15, K = kb..kb+7 and kb+16..kb+23, kb=8*(L>>4)
// B (32x16 KxN): lane L: n=L&15, K = kb..kb+7 and kb+16..kb+23, kb=8*(L>>4)
// C/D (16x16 f32): lane = N (L&15), VGPR r = M row r + 8*(L>>4)
// ---------------------------------------------------------------------------
__device__ __forceinline__ h16x16 fragA_lds(const _Float16* tileA, int wm, int lane) {
  int m  = (wm << 4) + (lane & 15);
  int kb = (lane >> 4) << 3;
  const _Float16* p = tileA + m * LDS_A_STRIDE + kb;
  h16x8 lo = *(const h16x8*)(p);
  h16x8 hi = *(const h16x8*)(p + 16);
  h16x16 r;
#pragma unroll
  for (int i = 0; i < 8; ++i) { r[i] = lo[i]; r[i + 8] = hi[i]; }
  return r;
}

__device__ __forceinline__ h16x16 fragB_lds(const _Float16* tileB, int nf, int lane) {
  const int n  = nf + (lane & 15);
  const int kb = (lane >> 4) << 3;
  h16x16 r;
#pragma unroll
  for (int i = 0; i < 8; ++i) {
    r[i]     = tileB[(kb + i) * LDS_B_STRIDE + n];
    r[i + 8] = tileB[(kb + 16 + i) * LDS_B_STRIDE + n];
  }
  return r;
}

__device__ __forceinline__ f32x8 wmma_f16(h16x16 a, h16x16 b, f32x8 c) {
  return __builtin_amdgcn_wmma_f32_16x16x32_f16(
      /*neg_a=*/false, a, /*neg_b=*/false, b,
      /*c_mod=*/(short)0, c, /*reuse_a=*/false, /*reuse_b=*/false);
}

// ---------------------------------------------------------------------------
// Data-prep kernels
// ---------------------------------------------------------------------------
__global__ void __launch_bounds__(256)
k_cvt_f32_to_f16(const float* __restrict__ in, _Float16* __restrict__ out, int n) {
  for (int i = blockIdx.x * blockDim.x + threadIdx.x; i < n;
       i += gridDim.x * blockDim.x)
    out[i] = (_Float16)in[i];
}

// in: [N][K] fp32 row-major -> out: [K][N] f16 row-major (B layout for WMMA)
__global__ void __launch_bounds__(256)
k_transpose_w(const float* __restrict__ in, _Float16* __restrict__ out,
              int N, int K) {
  int n = N * K;
  for (int i = blockIdx.x * blockDim.x + threadIdx.x; i < n;
       i += gridDim.x * blockDim.x) {
    int r = i / K, k = i - r * K;
    out[(size_t)k * N + r] = (_Float16)in[i];
  }
}

// enc [B][SRC][H] fp32 -> encT [B][H][SRC] f16
__global__ void __launch_bounds__(256)
k_transpose_enc(const float* __restrict__ enc, _Float16* __restrict__ encT) {
  const int n = B_ * H_ * SRC_;
  for (int i = blockIdx.x * blockDim.x + threadIdx.x; i < n;
       i += gridDim.x * blockDim.x) {
    int b   = i / (H_ * SRC_);
    int rem = i - b * (H_ * SRC_);
    int h   = rem / SRC_;
    int s   = rem - h * SRC_;
    encT[i] = (_Float16)enc[(size_t)b * SRC_ * H_ + (size_t)s * H_ + h];
  }
}

// ---------------------------------------------------------------------------
// GEMM 1: Q = dec · W_inT   (M=B*TGT, N=H, K=H), f16 out
// ---------------------------------------------------------------------------
__global__ void __launch_bounds__(256)
k_gemm_q(const _Float16* __restrict__ A, const _Float16* __restrict__ Bt,
         _Float16* __restrict__ C) {
  __shared__ _Float16 Abuf[2][64 * LDS_A_STRIDE];
  __shared__ _Float16 Bbuf[2][32 * LDS_B_STRIDE];
  const int tid = threadIdx.x, lane = tid & 31, w = tid >> 5;
  const int wm = w & 3, wn = w >> 2;
  const int m0 = blockIdx.y * 64, n0 = blockIdx.x * 128;
  f32x8 acc[4] = {};

  stage_A_f16(Abuf[0], A + (size_t)m0 * H_, H_, tid);
  stage_B_f16(Bbuf[0], Bt + n0, H_, tid);
  int buf = 0;
  for (int k0 = 0; k0 < H_; k0 += 32) {
    wait_async_and_barrier();
    if (k0 + 32 < H_) {
      stage_A_f16(Abuf[buf ^ 1], A + (size_t)m0 * H_ + k0 + 32, H_, tid);
      stage_B_f16(Bbuf[buf ^ 1], Bt + (size_t)(k0 + 32) * H_ + n0, H_, tid);
    }
    h16x16 a = fragA_lds(Abuf[buf], wm, lane);
#pragma unroll
    for (int j = 0; j < 4; ++j) {
      h16x16 b = fragB_lds(Bbuf[buf], wn * 64 + j * 16, lane);
      acc[j] = wmma_f16(a, b, acc[j]);
    }
    buf ^= 1;
  }
  const int n = lane & 15, mhi = (lane >> 4) * 8;
#pragma unroll
  for (int j = 0; j < 4; ++j)
#pragma unroll
    for (int r = 0; r < 8; ++r)
      C[(size_t)(m0 + wm * 16 + r + mhi) * H_ + n0 + wn * 64 + 16 * j + n] =
          (_Float16)acc[j][r];
}

// ---------------------------------------------------------------------------
// GEMM 2: scores = Q · encT per batch (M=TGT, N=SRC, K=H), fp32 -> align region
// ---------------------------------------------------------------------------
__global__ void __launch_bounds__(256)
k_gemm_align(const _Float16* __restrict__ Q, const _Float16* __restrict__ EncT,
             float* __restrict__ out_align) {
  __shared__ _Float16 Abuf[2][64 * LDS_A_STRIDE];
  __shared__ _Float16 Bbuf[2][32 * LDS_B_STRIDE];
  const int bat = blockIdx.z;
  const _Float16* A  = Q    + (size_t)bat * TGT_ * H_;
  const _Float16* Bt = EncT + (size_t)bat * H_ * SRC_;
  const int tid = threadIdx.x, lane = tid & 31, w = tid >> 5;
  const int wm = w & 3, wn = w >> 2;
  const int m0 = blockIdx.y * 64, n0 = blockIdx.x * 128;
  f32x8 acc[4] = {};

  stage_A_f16(Abuf[0], A + (size_t)m0 * H_, H_, tid);
  stage_B_f16(Bbuf[0], Bt + n0, SRC_, tid);
  int buf = 0;
  for (int k0 = 0; k0 < H_; k0 += 32) {
    wait_async_and_barrier();
    if (k0 + 32 < H_) {
      stage_A_f16(Abuf[buf ^ 1], A + (size_t)m0 * H_ + k0 + 32, H_, tid);
      stage_B_f16(Bbuf[buf ^ 1], Bt + (size_t)(k0 + 32) * SRC_ + n0, SRC_, tid);
    }
    h16x16 a = fragA_lds(Abuf[buf], wm, lane);
#pragma unroll
    for (int j = 0; j < 4; ++j) {
      h16x16 b = fragB_lds(Bbuf[buf], wn * 64 + j * 16, lane);
      acc[j] = wmma_f16(a, b, acc[j]);
    }
    buf ^= 1;
  }
  const int n = lane & 15, mhi = (lane >> 4) * 8;
#pragma unroll
  for (int j = 0; j < 4; ++j)
#pragma unroll
    for (int r = 0; r < 8; ++r) {
      int t = m0 + wm * 16 + r + mhi;
      out_align[(size_t)t * (B_ * SRC_) + (size_t)bat * SRC_ +
                n0 + wn * 64 + 16 * j + n] = acc[j][r];
    }
}

// ---------------------------------------------------------------------------
// Masked softmax over SRC, in place on out_align; one block per (t, b) row
// ---------------------------------------------------------------------------
__global__ void __launch_bounds__(256)
k_softmax(float* __restrict__ align, const int* __restrict__ lens) {
  const int t = blockIdx.x, b = blockIdx.y, tid = threadIdx.x;
  float* row = align + (size_t)t * (B_ * SRC_) + (size_t)b * SRC_;
  const int len = lens[b];
  __shared__ float red[256];

  float lmax = -1e30f;
  for (int s = tid; s < SRC_; s += 256) {
    float v = row[s];
    if (s < len) lmax = fmaxf(lmax, v);
  }
  red[tid] = lmax;
  __syncthreads();
  for (int off = 128; off > 0; off >>= 1) {
    if (tid < off) red[tid] = fmaxf(red[tid], red[tid + off]);
    __syncthreads();
  }
  const float bmax = red[0];
  __syncthreads();

  float lsum = 0.f;
  for (int s = tid; s < SRC_; s += 256)
    if (s < len) lsum += __expf(row[s] - bmax);
  red[tid] = lsum;
  __syncthreads();
  for (int off = 128; off > 0; off >>= 1) {
    if (tid < off) red[tid] += red[tid + off];
    __syncthreads();
  }
  const float inv = 1.f / red[0];

  for (int s = tid; s < SRC_; s += 256)
    row[s] = (s < len) ? __expf(row[s] - bmax) * inv : 0.f;
}

// ---------------------------------------------------------------------------
// GEMM 3: context = P · enc per batch (M=TGT, N=H, K=SRC), f16 out
// A staged with fused f32->f16 conversion; B staged async.
// ---------------------------------------------------------------------------
__global__ void __launch_bounds__(256)
k_gemm_ctx(const float* __restrict__ Pall, const _Float16* __restrict__ Enc,
           _Float16* __restrict__ Ctx) {
  __shared__ _Float16 Abuf[2][64 * LDS_A_STRIDE];
  __shared__ _Float16 Bbuf[2][32 * LDS_B_STRIDE];
  const int bat = blockIdx.z;
  const float* A = Pall + (size_t)bat * SRC_;          // row t stride B_*SRC_
  const _Float16* Bm = Enc + (size_t)bat * SRC_ * H_;  // [SRC][H] == [K][N]
  const int tid = threadIdx.x, lane = tid & 31, w = tid >> 5;
  const int wm = w & 3, wn = w >> 2;
  const int m0 = blockIdx.y * 64, n0 = blockIdx.x * 128;
  const long ldA = (long)B_ * SRC_;
  f32x8 acc[4] = {};

  stage_A_f32(Abuf[0], A + (long)m0 * ldA, ldA, tid);
  stage_B_f16(Bbuf[0], Bm + n0, H_, tid);
  int buf = 0;
  for (int k0 = 0; k0 < SRC_; k0 += 32) {
    wait_async_and_barrier();
    if (k0 + 32 < SRC_) {
      stage_A_f32(Abuf[buf ^ 1], A + (long)m0 * ldA + k0 + 32, ldA, tid);
      stage_B_f16(Bbuf[buf ^ 1], Bm + (size_t)(k0 + 32) * H_ + n0, H_, tid);
    }
    h16x16 a = fragA_lds(Abuf[buf], wm, lane);
#pragma unroll
    for (int j = 0; j < 4; ++j) {
      h16x16 b = fragB_lds(Bbuf[buf], wn * 64 + j * 16, lane);
      acc[j] = wmma_f16(a, b, acc[j]);
    }
    buf ^= 1;
  }
  const int n = lane & 15, mhi = (lane >> 4) * 8;
#pragma unroll
  for (int j = 0; j < 4; ++j)
#pragma unroll
    for (int r = 0; r < 8; ++r)
      Ctx[(size_t)(bat * TGT_ + m0 + wm * 16 + r + mhi) * H_ +
          n0 + wn * 64 + 16 * j + n] = (_Float16)acc[j][r];
}

// ---------------------------------------------------------------------------
// GEMM 4: attn = tanh([ctx | dec] · W_outT) (M=B*TGT, N=H, K=2H)
// fused output transpose to [t][b][h]
// ---------------------------------------------------------------------------
__global__ void __launch_bounds__(256)
k_gemm_out(const _Float16* __restrict__ Ctx, const _Float16* __restrict__ Dec,
           const _Float16* __restrict__ WoutT, float* __restrict__ out_attn) {
  __shared__ _Float16 Abuf[2][64 * LDS_A_STRIDE];
  __shared__ _Float16 Bbuf[2][32 * LDS_B_STRIDE];
  const int tid = threadIdx.x, lane = tid & 31, w = tid >> 5;
  const int wm = w & 3, wn = w >> 2;
  const int m0 = blockIdx.y * 64, n0 = blockIdx.x * 128;
  f32x8 acc[4] = {};

  stage_A_f16(Abuf[0], Ctx + (size_t)m0 * H_, H_, tid);
  stage_B_f16(Bbuf[0], WoutT + n0, H_, tid);
  int buf = 0;
  for (int k0 = 0; k0 < H2_; k0 += 32) {
    wait_async_and_barrier();
    if (k0 + 32 < H2_) {
      int kn = k0 + 32;
      const _Float16* Abase = (kn < H_) ? (Ctx + (size_t)m0 * H_ + kn)
                                        : (Dec + (size_t)m0 * H_ + (kn - H_));
      stage_A_f16(Abuf[buf ^ 1], Abase, H_, tid);
      stage_B_f16(Bbuf[buf ^ 1], WoutT + (size_t)kn * H_ + n0, H_, tid);
    }
    h16x16 a = fragA_lds(Abuf[buf], wm, lane);
#pragma unroll
    for (int j = 0; j < 4; ++j) {
      h16x16 b = fragB_lds(Bbuf[buf], wn * 64 + j * 16, lane);
      acc[j] = wmma_f16(a, b, acc[j]);
    }
    buf ^= 1;
  }
  const int n = lane & 15, mhi = (lane >> 4) * 8;
#pragma unroll
  for (int j = 0; j < 4; ++j)
#pragma unroll
    for (int r = 0; r < 8; ++r) {
      int m = m0 + wm * 16 + r + mhi;
      int t = m & (TGT_ - 1);
      int b = m >> 9;  // TGT_ == 512
      out_attn[(size_t)t * (B_ * H_) + (size_t)b * H_ + n0 + wn * 64 + 16 * j + n] =
          tanhf(acc[j][r]);
    }
}

// ---------------------------------------------------------------------------
// Host-side orchestration
// ---------------------------------------------------------------------------
extern "C" void kernel_launch(void* const* d_in, const int* in_sizes, int n_in,
                              void* d_out, int out_size, void* d_ws, size_t ws_size,
                              hipStream_t stream) {
  (void)in_sizes; (void)n_in; (void)out_size; (void)ws_size;

  const float* dec   = (const float*)d_in[0];  // [B,TGT,H]
  const float* enc   = (const float*)d_in[1];  // [B,SRC,H]
  const int*   lens  = (const int*)d_in[2];    // [B]
  const float* w_in  = (const float*)d_in[3];  // [H,H]
  const float* w_out = (const float*)d_in[4];  // [H,2H]

  float* out_attn  = (float*)d_out;                           // [TGT,B,H]
  float* out_align = (float*)d_out + (size_t)TGT_ * B_ * H_;  // [TGT,B,SRC]

  // workspace layout (f16 tensors)
  char* ws = (char*)d_ws;
  _Float16* dec_h   = (_Float16*)(ws);                         size_t o = (size_t)B_*TGT_*H_*2;
  _Float16* enc_h   = (_Float16*)(ws + o);  o += (size_t)B_*SRC_*H_*2;
  _Float16* encT_h  = (_Float16*)(ws + o);  o += (size_t)B_*H_*SRC_*2;
  _Float16* winT_h  = (_Float16*)(ws + o);  o += (size_t)H_*H_*2;
  _Float16* woutT_h = (_Float16*)(ws + o);  o += (size_t)H2_*H_*2;
  _Float16* q_h     = (_Float16*)(ws + o);  o += (size_t)B_*TGT_*H_*2;
  _Float16* ctx_h   = (_Float16*)(ws + o);

  const dim3 blk(256);

  // 1) precision conversion + operand transposes
  k_cvt_f32_to_f16<<<dim3(2048), blk, 0, stream>>>(dec, dec_h, B_ * TGT_ * H_);
  k_cvt_f32_to_f16<<<dim3(4096), blk, 0, stream>>>(enc, enc_h, B_ * SRC_ * H_);
  k_transpose_enc<<<dim3(4096), blk, 0, stream>>>(enc, encT_h);
  k_transpose_w<<<dim3(512), blk, 0, stream>>>(w_in, winT_h, H_, H_);
  k_transpose_w<<<dim3(1024), blk, 0, stream>>>(w_out, woutT_h, H_, H2_);

  // 2) Q = dec · W_inT
  k_gemm_q<<<dim3(H_ / 128, (B_ * TGT_) / 64), blk, 0, stream>>>(dec_h, winT_h, q_h);

  // 3) scores -> d_out align region
  k_gemm_align<<<dim3(SRC_ / 128, TGT_ / 64, B_), blk, 0, stream>>>(q_h, encT_h, out_align);

  // 4) masked softmax in place
  k_softmax<<<dim3(TGT_, B_), blk, 0, stream>>>(out_align, lens);

  // 5) context = P · enc
  k_gemm_ctx<<<dim3(H_ / 128, TGT_ / 64, B_), blk, 0, stream>>>(out_align, enc_h, ctx_h);

  // 6) attn = tanh([ctx|dec] · W_outT) -> d_out attn region
  k_gemm_out<<<dim3(H_ / 128, (B_ * TGT_) / 64), blk, 0, stream>>>(ctx_h, dec_h, woutT_h, out_attn);
}
